// unetUp_2_33947421508451
// MI455X (gfx1250) — matrix operand order repaired and checked by reading, not compiled, not run on hardware
//
#include <hip/hip_runtime.h>
#include <hip/hip_bf16.h>
#include <math.h>

typedef __attribute__((ext_vector_type(16))) _Float16 v16h;
typedef __attribute__((ext_vector_type(8)))  _Float16 v8h;
typedef __attribute__((ext_vector_type(8)))  float    v8f;

// ---------------------------------------------------------------------------
// Weight pack: fp32 [M][K] -> f16 [Mp][K], zero padded rows (K already %32==0)
// ---------------------------------------------------------------------------
__global__ void pack_w_kernel(const float* __restrict__ src, _Float16* __restrict__ dst,
                              int M, int K, int Mp) {
    int idx = blockIdx.x * blockDim.x + threadIdx.x;
    if (idx >= Mp * K) return;
    int m = idx / K, k = idx - m * K;
    float v = (m < M) ? src[m * K + k] : 0.0f;
    dst[idx] = (_Float16)v;
}

__global__ void f32_to_f16_kernel(const float* __restrict__ src, _Float16* __restrict__ dst, int n) {
    int i = blockIdx.x * blockDim.x + threadIdx.x;
    if (i < n) dst[i] = (_Float16)src[i];
}

__global__ void zero_f16_kernel(_Float16* __restrict__ dst, int n) {
    int i = blockIdx.x * blockDim.x + threadIdx.x;
    if (i < n) dst[i] = (_Float16)0.0f;
}

// ---------------------------------------------------------------------------
// Implicit-im2col conv as WMMA GEMM with LDS-staged B panel.
//   Inputs for KS==3 are halo-padded [B][CIN][H+2][W+2] (zero halo) so the
//   im2col gather is unconditional. 8 waves/block share one 64-pixel strip
//   (same b,y) and cover 8 M-tiles; the block stages the im2col panel
//   [64 px][KC] in LDS once per K-chunk, then each wave reads B fragments as
//   aligned 2x ds_load_b128 and A fragments as 2x 16B global loads.
// Fragment layouts used by this kernel (wave32, contiguous K for A and B):
//   A: lanes 0-15 M=lane, halves = K0..15; lanes 16-31 K16..31
//   B: lane = N col; 16 halves = 16 consecutive K (lo lanes K0..15, hi +16)
//   C/D: vgpr r: M = r + (hi?8:0), N = lane&15
// ---------------------------------------------------------------------------
template<int KS, int H, int W, int CIN, int KC, int OPAD>
__global__ __launch_bounds__(256)
void wmma_conv_kernel(const _Float16* __restrict__ Wt,
                      const _Float16* __restrict__ Xin,
                      const float* __restrict__ bias,
                      float* __restrict__ outF,
                      _Float16* __restrict__ outH,
                      int M, int Mp, int relu, int Cstore) {
    constexpr int IPAD = KS / 2;
    constexpr int Hp = H + 2 * IPAD, Wp = W + 2 * IPAD;   // input (padded) dims
    constexpr int Ho = H + 2 * OPAD, Wo = W + 2 * OPAD;   // output (padded) dims
    constexpr int K   = CIN * KS * KS;                    // multiple of KC
    constexpr int ROW = KC + 16;                          // LDS row stride (32B mult)

    __shared__ __align__(32) _Float16 smem[64 * ROW];

    const int tid  = threadIdx.x;
    const int lane = tid & 31;
    const int wave = tid >> 5;
    const int li   = lane & 15;
    const bool hi  = lane >= 16;

    const int n0 = blockIdx.x * 64;            // 64-pixel strip, same (b,y)
    const int b  = n0 / (H * W);
    const int y  = (n0 / W) % H;
    const int x0 = n0 % W;

    const int m0   = wave * 16;                          // wave's M tile
    const int mrow = (m0 < Mp) ? m0 : (Mp - 16);         // clamp idle waves

    v8f acc[4] = {};
    for (int kc = 0; kc < K; kc += KC) {
        __syncthreads();
        // ---- cooperative im2col staging: [p=0..63][k=0..KC-1] ----
        constexpr int E = 64 * KC;
#pragma unroll 4
        for (int e = tid; e < E; e += 256) {
            int p = e / KC;                  // constexpr divisor
            int k = e - p * KC;
            int kg = kc + k;
            int ci, q, dxr;
            if (KS == 1) { ci = kg; q = 0; dxr = 0; }
            else {
                ci = kg / 9;                 // magic multiply
                int r = kg - 9 * ci;
                q = r / 3;                   // 0..2  (row offset in padded coords)
                dxr = r - 3 * q;             // 0..2  (col offset in padded coords)
            }
            smem[p * ROW + k] = Xin[((b * CIN + ci) * Hp + y + q) * Wp + x0 + p + dxr];
        }
        __syncthreads();

        // ---- GEMM over this K-chunk ----
        for (int k0 = 0; k0 < KC; k0 += 32) {
            const _Float16* wrow = Wt + (mrow + li) * K + kc + k0 + (hi ? 16 : 0);
            const v8h* wp = (const v8h*)wrow;
            v8h alo = wp[0];                 // K k..k+7
            v8h ahi = wp[1];                 // K k+8..k+15
            v16h a;
#pragma unroll
            for (int j = 0; j < 8; ++j) { a[j] = alo[j]; a[j + 8] = ahi[j]; }

#pragma unroll
            for (int t = 0; t < 4; ++t) {
                const int p = t * 16 + li;
                v16h bf = *(const v16h*)(&smem[p * ROW + k0 + (hi ? 16 : 0)]);
                acc[t] = __builtin_amdgcn_wmma_f32_16x16x32_f16(false, a, false, bf,
                                                                (short)0, acc[t], false, false);
            }
        }
    }

    // ---- store ----
    const int mh = mrow + (hi ? 8 : 0);
#pragma unroll
    for (int t = 0; t < 4; ++t) {
        const int xs = x0 + t * 16 + li;
#pragma unroll
        for (int r = 0; r < 8; ++r) {
            int m = mh + r;
            if (m0 < Mp && m < M) {
                float v = acc[t][r] + (bias ? bias[m] : 0.0f);
                if (relu) v = fmaxf(v, 0.0f);
                int idx = ((b * Cstore + m) * Ho + y + OPAD) * Wo + xs + OPAD;
                if (outF) outF[idx] = v; else outH[idx] = (_Float16)v;
            }
        }
    }
}

// ---------------------------------------------------------------------------
// BN batch statistics: one block per channel, reduce over Bn*HW (fp32, unpadded)
// ---------------------------------------------------------------------------
__global__ void bn_stats_kernel(const float* __restrict__ x, float* __restrict__ stats,
                                int C, int HWp, int Bn) {
    __shared__ float ss[256], sq[256];
    int c = blockIdx.x;
    int n = Bn * HWp;
    float s = 0.0f, q = 0.0f;
    for (int i = threadIdx.x; i < n; i += blockDim.x) {
        int b = i / HWp, p = i - b * HWp;
        float v = x[(b * C + c) * HWp + p];
        s += v; q += v * v;
    }
    ss[threadIdx.x] = s; sq[threadIdx.x] = q;
    __syncthreads();
    for (int o = 128; o > 0; o >>= 1) {
        if (threadIdx.x < o) {
            ss[threadIdx.x] += ss[threadIdx.x + o];
            sq[threadIdx.x] += sq[threadIdx.x + o];
        }
        __syncthreads();
    }
    if (threadIdx.x == 0) {
        float mean = ss[0] / n;
        float var  = sq[0] / n - mean * mean;
        stats[c * 2]     = mean;
        stats[c * 2 + 1] = rsqrtf(var + 1e-5f);
    }
}

// BN apply + ReLU, fp32 [B][C][64][64] -> f16 halo-padded [B][C][66][66]
__global__ void bn_relu_pad_kernel(const float* __restrict__ x, const float* __restrict__ stats,
                                   const float* __restrict__ g, const float* __restrict__ bt,
                                   _Float16* __restrict__ out, int C, int n) {
    int i = blockIdx.x * blockDim.x + threadIdx.x;
    if (i >= n) return;
    int xcol = i & 63, yrow = (i >> 6) & 63, bc = i >> 12;
    int c = bc % C;
    float v = (x[i] - stats[c * 2]) * stats[c * 2 + 1] * g[c] + bt[c];
    out[(bc * 66 + yrow + 1) * 66 + xcol + 1] = (_Float16)fmaxf(v, 0.0f);
}

// ---------------------------------------------------------------------------
// pixel_shuffle(r=2) + BN affine + softmax over 25 kernel weights, fused.
// ---------------------------------------------------------------------------
__global__ void shuffle_softmax_kernel(const float* __restrict__ Y2, const float* __restrict__ stats,
                                       const float* __restrict__ g, const float* __restrict__ bt,
                                       float* __restrict__ Wsm) {
    int i = blockIdx.x * blockDim.x + threadIdx.x;
    if (i >= 4 * 128 * 128) return;
    int ox = i & 127, oy = (i >> 7) & 127, b = i >> 14;
    int ri = oy & 1, rj = ox & 1, y = oy >> 1, x = ox >> 1;
    float v[25], mx = -1e30f;
#pragma unroll
    for (int co = 0; co < 25; ++co) {
        int ch = co * 4 + ri * 2 + rj;
        float t = (Y2[(b * 100 + ch) * 4096 + y * 64 + x] - stats[ch * 2])
                  * stats[ch * 2 + 1] * g[ch] + bt[ch];
        v[co] = t; mx = fmaxf(mx, t);
    }
    float s = 0.0f;
#pragma unroll
    for (int co = 0; co < 25; ++co) { v[co] = __expf(v[co] - mx); s += v[co]; }
    float inv = 1.0f / s;
#pragma unroll
    for (int co = 0; co < 25; ++co)
        Wsm[(b * 25 + co) * 16384 + oy * 128 + ox] = v[co] * inv;
}

// ---------------------------------------------------------------------------
// Bilinear 2x upsample (half-pixel centers, edge clamp), fp32 -> f16
// ---------------------------------------------------------------------------
__global__ void bilinear_kernel(const float* __restrict__ X, _Float16* __restrict__ Xup, int C) {
    int i = blockIdx.x * blockDim.x + threadIdx.x;
    if (i >= 4 * C * 128 * 128) return;
    int ox = i & 127, oy = (i >> 7) & 127, bc = i >> 14;
    float fy = oy * 0.5f - 0.25f, fx = ox * 0.5f - 0.25f;
    int y0 = (int)floorf(fy), x0 = (int)floorf(fx);
    float wy = fy - y0, wx = fx - x0;
    int y0c = max(0, min(63, y0)), y1c = max(0, min(63, y0 + 1));
    int x0c = max(0, min(63, x0)), x1c = max(0, min(63, x0 + 1));
    const float* p = X + bc * 4096;
    float v = (1.0f - wy) * ((1.0f - wx) * p[y0c * 64 + x0c] + wx * p[y0c * 64 + x1c])
            +          wy * ((1.0f - wx) * p[y1c * 64 + x0c] + wx * p[y1c * 64 + x1c]);
    Xup[i] = (_Float16)v;
}

// ---------------------------------------------------------------------------
// CARAFE apply: one block per (b, pixel); 256 threads = 256 channels;
// the 25 softmax weights staged in LDS.
// ---------------------------------------------------------------------------
__global__ __launch_bounds__(256)
void carafe_kernel(const float* __restrict__ Wsm, const _Float16* __restrict__ Xup,
                   _Float16* __restrict__ Up) {
    __shared__ float w[25];
    int pos = blockIdx.x & 16383;
    int b   = blockIdx.x >> 14;
    int ox  = pos & 127, oy = pos >> 7;
    if (threadIdx.x < 25) w[threadIdx.x] = Wsm[(b * 25 + threadIdx.x) * 16384 + pos];
    __syncthreads();
    int c = threadIdx.x;
    const _Float16* src = Xup + (b * 256 + c) * 16384;
    float acc = 0.0f;
#pragma unroll
    for (int ii = 0; ii < 5; ++ii) {
        int yy = oy + 2 * ii - 4;
        if ((unsigned)yy < 128u) {
#pragma unroll
            for (int jj = 0; jj < 5; ++jj) {
                int xx = ox + 2 * jj - 4;
                if ((unsigned)xx < 128u)
                    acc += w[ii * 5 + jj] * (float)src[yy * 128 + xx];
            }
        }
    }
    Up[(b * 256 + c) * 16384 + pos] = (_Float16)acc;
}

// low_feature0 fp32 [4,128,128,128] -> channels 128..255 of padded Cat
// f16 [4,256,130,130]
__global__ void concat_low_kernel(const float* __restrict__ low, _Float16* __restrict__ Catp) {
    int i = blockIdx.x * blockDim.x + threadIdx.x;
    if (i >= 4 * 128 * 16384) return;
    int pos = i & 16383; int x = pos & 127; int y = pos >> 7;
    int t = i >> 14; int c = t & 127; int b = t >> 7;
    Catp[((b * 256 + 128 + c) * 130 + y + 1) * 130 + x + 1] = (_Float16)low[i];
}

// ---------------------------------------------------------------------------
extern "C" void kernel_launch(void* const* d_in, const int* in_sizes, int n_in,
                              void* d_out, int out_size, void* d_ws, size_t ws_size,
                              hipStream_t stream) {
    (void)in_sizes; (void)n_in; (void)out_size; (void)ws_size;
    const float* high   = (const float*)d_in[0];
    const float* low    = (const float*)d_in[1];
    const float* comp_w = (const float*)d_in[2];
    const float* bn1_g  = (const float*)d_in[4];
    const float* bn1_b  = (const float*)d_in[5];
    const float* enc_w  = (const float*)d_in[6];
    const float* bn2_g  = (const float*)d_in[8];
    const float* bn2_b  = (const float*)d_in[9];
    const float* conv1_w = (const float*)d_in[10];
    const float* conv1_b = (const float*)d_in[11];
    const float* c2w1 = (const float*)d_in[12];
    const float* c2b1 = (const float*)d_in[13];
    const float* c2w2 = (const float*)d_in[14];
    const float* c2b2 = (const float*)d_in[15];
    float* Out = (float*)d_out;

    // Workspace layout (liveness-based reuse, peak ~96MB; fits in the 192MB L2):
    char* ws = (char*)d_ws;
    const size_t MB = 1u << 20;
    _Float16* WT    = (_Float16*)ws;                // packed f16 weights ~1.1MB
    _Float16* Wcomp = WT;                           //  64 x  256
    _Float16* Wenc  = WT + 16384;                   // 112 x  576 (M=100)
    _Float16* Wc1   = WT + 80896;                   // 128 x  256
    _Float16* Wc21  = WT + 113664;                  // 128 x 2304
    _Float16* Wc22  = WT + 408576;                  // 128 x 1152
    float*    st1   = (float*)(ws + 2 * MB);        // 64 ch stats
    float*    st2   = st1 + 256;                    // 100 ch stats
    _Float16* Xh    = (_Float16*)(ws + 3 * MB);     //  8MB, dead after comp conv
    float*    Y1    = (float*)(ws + 11 * MB);       //  4MB, dead after BN1 apply
    _Float16* Y1hp  = (_Float16*)(ws + 15 * MB);    //  2.3MB padded, dead after enc
    float*    Y2    = (float*)(ws + 18 * MB);       //  6.6MB, dead after softmax
    float*    Wsm   = (float*)(ws + 25 * MB);       //  6.6MB, dead after carafe
    _Float16* Xup   = (_Float16*)(ws + 32 * MB);    // 32MB, dead after carafe
    _Float16* Up    = (_Float16*)(ws + 64 * MB);    // 32MB, dead after conv1
    _Float16* Catp  = (_Float16*)(ws + 3 * MB);     // 34.6MB padded, reuses 3..38
    _Float16* Act2p = (_Float16*)(ws + 38 * MB);    // 17.3MB padded, reuses 38..56

    auto cdiv = [](int a, int b) { return (a + b - 1) / b; };

    // Pack weights to f16 [Mp][K]
    pack_w_kernel<<<cdiv(64 * 256, 256),   256, 0, stream>>>(comp_w,  Wcomp, 64, 256, 64);
    pack_w_kernel<<<cdiv(112 * 576, 256),  256, 0, stream>>>(enc_w,   Wenc, 100, 576, 112);
    pack_w_kernel<<<cdiv(128 * 256, 256),  256, 0, stream>>>(conv1_w, Wc1, 128, 256, 128);
    pack_w_kernel<<<cdiv(128 * 2304, 256), 256, 0, stream>>>(c2w1,    Wc21, 128, 2304, 128);
    pack_w_kernel<<<cdiv(128 * 1152, 256), 256, 0, stream>>>(c2w2,    Wc22, 128, 1152, 128);
    f32_to_f16_kernel<<<cdiv(4 * 256 * 4096, 256), 256, 0, stream>>>(high, Xh, 4 * 256 * 4096);
    zero_f16_kernel<<<cdiv(4 * 64 * 66 * 66, 256), 256, 0, stream>>>(Y1hp, 4 * 64 * 66 * 66);

    // comp 1x1: M=64,K=256,N=16384 (bias absorbed by BN mean). 16384/64 strips
    wmma_conv_kernel<1, 64, 64, 256, 256, 0><<<dim3(256), 256, 0, stream>>>(
        Wcomp, Xh, nullptr, Y1, nullptr, 64, 64, 0, 64);
    bn_stats_kernel<<<64, 256, 0, stream>>>(Y1, st1, 64, 4096, 4);
    bn_relu_pad_kernel<<<cdiv(1048576, 256), 256, 0, stream>>>(Y1, st1, bn1_g, bn1_b,
                                                               Y1hp, 64, 1048576);
    // enc 3x3: M=100 (Mp=112), K=576, KC=192
    wmma_conv_kernel<3, 64, 64, 64, 192, 0><<<dim3(256), 256, 0, stream>>>(
        Wenc, Y1hp, nullptr, Y2, nullptr, 100, 112, 0, 100);
    bn_stats_kernel<<<100, 256, 0, stream>>>(Y2, st2, 100, 4096, 4);
    shuffle_softmax_kernel<<<cdiv(65536, 256), 256, 0, stream>>>(Y2, st2, bn2_g, bn2_b, Wsm);

    bilinear_kernel<<<cdiv(4 * 256 * 16384, 256), 256, 0, stream>>>(high, Xup, 256);
    carafe_kernel<<<65536, 256, 0, stream>>>(Wsm, Xup, Up);

    // Padded concat/act buffers become live only now (aliased regions are dead)
    zero_f16_kernel<<<cdiv(4 * 256 * 130 * 130, 256), 256, 0, stream>>>(Catp, 4 * 256 * 130 * 130);
    zero_f16_kernel<<<cdiv(4 * 128 * 130 * 130, 256), 256, 0, stream>>>(Act2p, 4 * 128 * 130 * 130);

    // conv1 1x1: M=128,K=256,N=65536 -> Catp channels 0..127 (Cstore=256, OPAD=1)
    wmma_conv_kernel<1, 128, 128, 256, 256, 1><<<dim3(1024), 256, 0, stream>>>(
        Wc1, Up, conv1_b, nullptr, Catp, 128, 128, 0, 256);
    concat_low_kernel<<<cdiv(4 * 128 * 16384, 256), 256, 0, stream>>>(low, Catp);

    // c2 conv1 3x3 + relu: M=128,K=2304 -> Act2p (OPAD=1)
    wmma_conv_kernel<3, 128, 128, 256, 256, 1><<<dim3(1024), 256, 0, stream>>>(
        Wc21, Catp, c2b1, nullptr, Act2p, 128, 128, 1, 128);
    // c2 conv2 3x3 + relu -> fp32 output (unpadded)
    wmma_conv_kernel<3, 128, 128, 128, 192, 0><<<dim3(1024), 256, 0, stream>>>(
        Wc22, Act2p, c2b2, Out, nullptr, 128, 128, 1, 128);
}
